// Cropr_21285857919420
// MI455X (gfx1250) — matrix operand order repaired and checked by reading, not compiled, not run on hardware
//
#include <hip/hip_runtime.h>
#include <hip/hip_bf16.h>
#include <math.h>

typedef __bf16 bf16_t;
typedef __attribute__((ext_vector_type(16))) __bf16 v16bf;
typedef __attribute__((ext_vector_type(8)))  __bf16 v8bf;
typedef __attribute__((ext_vector_type(8)))  float  v8f;
typedef __attribute__((ext_vector_type(4)))  int    v4i;

static constexpr int Bsz = 4;
static constexpr int Nn  = 2049;
static constexpr int Dd  = 1024;
static constexpr int Mm  = 1024;
static constexpr int Cc  = 80;
static constexpr int Hh  = 16;
static constexpr int HDd = 64;
static constexpr int HID = 4096;
static constexpr int PRUNE = 40;
static constexpr int NKEEP = Nn - PRUNE;       // 2009
static constexpr int XB_ROWS = 2080;           // zero-padded key rows (covers 32-key tail)
static constexpr int XT_LD   = 2080;           // zero-padded key stride in x^T
static constexpr int HEAD_NPAD = 128;          // head weight rows padded with zeros
static constexpr int ROWS  = Bsz * Mm;         // 4096 flattened (b, m) rows

// ---- d_out element offsets (fp32/int32 units, return-order concat) ----
static constexpr size_t O_XKEEP   = 0;
static constexpr size_t O_XR      = O_XKEEP   + (size_t)Bsz * NKEEP     * Dd;
static constexpr size_t O_POSKEEP = O_XR      + (size_t)Bsz * PRUNE     * Dd;
static constexpr size_t O_POSR    = O_POSKEEP + (size_t)Bsz * (NKEEP-1) * Dd;
static constexpr size_t O_IDXKEEP = O_POSR    + (size_t)Bsz * PRUNE     * Dd;
static constexpr size_t O_IDXR    = O_IDXKEEP + (size_t)Bsz * NKEEP;
static constexpr size_t O_PRED    = O_IDXR    + (size_t)Bsz * PRUNE;

// -------------------------------------------------------------------------
// Async global->LDS staging (CDNA5 ASYNCcnt path) with safe fallback
// -------------------------------------------------------------------------
#if __has_builtin(__builtin_amdgcn_global_load_async_to_lds_b128)
#define HAVE_ASYNC_LDS 1
typedef __attribute__((address_space(1))) v4i as1_v4i;
typedef __attribute__((address_space(3))) v4i as3_v4i;
__device__ __forceinline__ void async_copy16(const bf16_t* g, bf16_t* l) {
  as1_v4i* gp = (as1_v4i*)(unsigned long long)(g);
  as3_v4i* lp = (as3_v4i*)(unsigned)(unsigned long long)(l);
  __builtin_amdgcn_global_load_async_to_lds_b128(gp, lp, 0, 0);
}
#else
#define HAVE_ASYNC_LDS 0
__device__ __forceinline__ void async_copy16(const bf16_t* g, bf16_t* l) {
  *reinterpret_cast<v8bf*>(l) = *reinterpret_cast<const v8bf*>(g);
}
#endif

__device__ __forceinline__ void async_wait_all() {
#if HAVE_ASYNC_LDS
#if __has_builtin(__builtin_amdgcn_s_wait_asynccnt)
  __builtin_amdgcn_s_wait_asynccnt(0);
#else
  asm volatile("s_wait_asynccnt 0x0" ::: "memory");
#endif
#endif
}

// -------------------------------------------------------------------------
// WMMA helpers
// -------------------------------------------------------------------------
__device__ __forceinline__ v8f wmma_bf16(v16bf a, v16bf b, v8f c) {
  return __builtin_amdgcn_wmma_f32_16x16x32_bf16(false, a, false, b, (short)0, c,
                                                 false, false);
}

// Unguarded A-matrix 16x32 bf16 tile (ISA layout: lane 0-15 = row, K {0..7,16..23};
// lane 16-31 = row, K {8..15,24..31}). Source row-major, leading dim ld.
__device__ __forceinline__ v16bf load_tile_a(const bf16_t* __restrict__ base, int ld,
                                             int row0, int col0, int lane) {
  int r = row0 + (lane & 15);
  int c = col0 + ((lane >> 4) << 3);
  const bf16_t* p = base + (size_t)r * (size_t)ld + c;
  v8bf lo = *reinterpret_cast<const v8bf*>(p);
  v8bf hi = *reinterpret_cast<const v8bf*>(p + 16);
  return __builtin_shufflevector(lo, hi, 0,1,2,3,4,5,6,7,8,9,10,11,12,13,14,15);
}

// Unguarded B-matrix 32x16 bf16 tile from B^T rows (row-major): lane 0-15 = col n,
// K {0..7,16..23}; lane 16-31 = col n, K {8..15,24..31} (same K split as A).
__device__ __forceinline__ v16bf load_tile_b(const bf16_t* __restrict__ baseT, int ld,
                                             int row0, int col0, int lane) {
  int r = row0 + (lane & 15);
  int c = col0 + ((lane >> 4) << 3);
  const bf16_t* p = baseT + (size_t)r * (size_t)ld + c;
  v8bf lo = *reinterpret_cast<const v8bf*>(p);
  v8bf hi = *reinterpret_cast<const v8bf*>(p + 16);
  return __builtin_shufflevector(lo, hi, 0,1,2,3,4,5,6,7,8,9,10,11,12,13,14,15);
}

// -------------------------------------------------------------------------
// Prep kernels: convert / pad / transpose-convert to bf16
// -------------------------------------------------------------------------
__global__ void cvt_kernel(const float* __restrict__ in, bf16_t* __restrict__ out,
                           size_t n) {
  size_t i = (size_t)blockIdx.x * blockDim.x + threadIdx.x;
  if (i < n) out[i] = (bf16_t)in[i];
}

// Per-batch convert with zero-padded rows: out (rowsPad x cols), in (rows x cols).
__global__ void cvt_pad_kernel(const float* __restrict__ in, bf16_t* __restrict__ out,
                               int rows, int rowsPad, int cols) {
  int bz = blockIdx.z;
  size_t i = (size_t)blockIdx.x * blockDim.x + threadIdx.x;
  size_t tot = (size_t)rowsPad * cols;
  if (i >= tot) return;
  int r = (int)(i / cols);
  int c = (int)(i % cols);
  const float* ib = in + (size_t)bz * rows * cols;
  bf16_t* ob = out + (size_t)bz * tot;
  ob[i] = (r < rows) ? (bf16_t)ib[(size_t)r * cols + c] : (bf16_t)0.0f;
}

// out[or, oc] = in[oc, or] (bf16); zero-filled for oc >= R or or >= C (padding).
__global__ void tr_cvt_kernel(const float* __restrict__ in, bf16_t* __restrict__ out,
                              int R, int C, int ldout, int outCols, int outRows,
                              size_t inStride, size_t outStride) {
  int oc  = blockIdx.x * 16 + threadIdx.x;
  int orw = blockIdx.y * 16 + threadIdx.y;
  int bz  = blockIdx.z;
  if (orw >= outRows || oc >= outCols) return;
  const float* ib = in + inStride * bz;
  bf16_t* ob = out + outStride * bz;
  float v = (oc < R && orw < C) ? ib[(size_t)oc * C + orw] : 0.0f;
  ob[(size_t)orw * ldout + oc] = (bf16_t)v;
}

// -------------------------------------------------------------------------
// Scores: scores[b,n] = (sum_m queries[m,:]) . x[b,n,:]
// -------------------------------------------------------------------------
__global__ void qsum_kernel(const float* __restrict__ q, float* __restrict__ qs) {
  int d = blockIdx.x * blockDim.x + threadIdx.x;
  if (d >= Dd) return;
  float s = 0.f;
  for (int m = 0; m < Mm; ++m) s += q[(size_t)m * Dd + d];
  qs[d] = s;
}

__global__ __launch_bounds__(256) void scores_kernel(const float* __restrict__ x,
                                                     const float* __restrict__ qs,
                                                     float* __restrict__ sc) {
  __shared__ float red[256];
  int n = blockIdx.x, b = blockIdx.y, tid = threadIdx.x;
  const float* xr = x + ((size_t)b * Nn + n) * Dd;
  float s = 0.f;
  for (int d = tid; d < Dd; d += 256) s += xr[d] * qs[d];
  red[tid] = s;
  __syncthreads();
  for (int off = 128; off > 0; off >>= 1) {
    if (tid < off) red[tid] += red[tid + off];
    __syncthreads();
  }
  if (tid == 0) sc[(size_t)b * Nn + n] = red[0];
}

// -------------------------------------------------------------------------
// Per-batch bitonic argsort (descending), N=2049 padded to 4096 in LDS
// -------------------------------------------------------------------------
__global__ __launch_bounds__(1024) void sort_kernel(const float* __restrict__ sc,
                                                    int* __restrict__ order) {
  __shared__ float skey[4096];
  __shared__ int   sval[4096];
  const int b = blockIdx.x, tid = threadIdx.x;
  for (int t = tid; t < 4096; t += 1024) {
    float k;
    if (t == 0)       k = __builtin_inff();
    else if (t < Nn)  k = sc[(size_t)b * Nn + t];
    else              k = -__builtin_inff();
    skey[t] = k;
    sval[t] = t;
  }
  __syncthreads();
  for (int k = 2; k <= 4096; k <<= 1) {
    for (int j = k >> 1; j > 0; j >>= 1) {
      for (int t = tid; t < 4096; t += 1024) {
        int ixj = t ^ j;
        if (ixj > t) {
          bool desc = ((t & k) == 0);
          float a = skey[t], c = skey[ixj];
          bool sw = desc ? (a < c) : (a > c);
          if (sw) {
            skey[t] = c; skey[ixj] = a;
            int tv = sval[t]; sval[t] = sval[ixj]; sval[ixj] = tv;
          }
        }
      }
      __syncthreads();
    }
  }
  for (int t = tid; t < Nn; t += 1024) order[(size_t)b * Nn + t] = sval[t];
}

// -------------------------------------------------------------------------
// Gathers
// -------------------------------------------------------------------------
__global__ __launch_bounds__(256) void gather_x_kernel(const float* __restrict__ x,
                                                       const int* __restrict__ order,
                                                       float* __restrict__ out) {
  int blk = blockIdx.x;
  int b = blk / Nn, i = blk % Nn;
  int src = order[(size_t)b * Nn + i];
  src = min(max(src, 0), Nn - 1);
  const float* s = x + ((size_t)b * Nn + src) * Dd;
  float* d;
  if (i < NKEEP) d = out + O_XKEEP + ((size_t)b * NKEEP + i) * Dd;
  else           d = out + O_XR    + ((size_t)b * PRUNE + (i - NKEEP)) * Dd;
  for (int c = threadIdx.x; c < Dd; c += 256) d[c] = s[c];
}

__global__ __launch_bounds__(256) void gather_pos_kernel(const float* __restrict__ pos,
                                                         const int* __restrict__ order,
                                                         float* __restrict__ out) {
  int blk = blockIdx.x;
  int b = blk / (Nn - 1), i = blk % (Nn - 1);
  int src = order[(size_t)b * Nn + i + 1] - 1;
  src = min(max(src, 0), Nn - 2);
  const float* s = pos + ((size_t)b * (Nn - 1) + src) * Dd;
  float* d;
  if (i < NKEEP - 1) d = out + O_POSKEEP + ((size_t)b * (NKEEP - 1) + i) * Dd;
  else               d = out + O_POSR    + ((size_t)b * PRUNE + (i - (NKEEP - 1))) * Dd;
  for (int c = threadIdx.x; c < Dd; c += 256) d[c] = s[c];
}

__global__ void gather_idx_kernel(const int* __restrict__ idx,
                                  const int* __restrict__ order,
                                  int* __restrict__ outi) {
  int t = blockIdx.x * blockDim.x + threadIdx.x;
  if (t >= Bsz * Nn) return;
  int b = t / Nn, i = t % Nn;
  int src = order[(size_t)b * Nn + i];
  src = min(max(src, 0), Nn - 1);
  int v = idx[(size_t)b * Nn + src];
  if (i < NKEEP) outi[O_IDXKEEP + (size_t)b * NKEEP + i] = v;
  else           outi[O_IDXR    + (size_t)b * PRUNE + (i - NKEEP)] = v;
}

// -------------------------------------------------------------------------
// Flash-style cross-attention: one wave per (b, h, 16-row m-tile).
// ctx[b,m,h*64+d] = softmax(Q K^T / 8) K, in bf16. All tile loads branch-free
// (key buffers zero-padded to XB_ROWS / XT_LD); softmax tail masked per lane.
// -------------------------------------------------------------------------
__global__ __launch_bounds__(32) void attn_kernel(const bf16_t* __restrict__ qb,
                                                  const bf16_t* __restrict__ xb,
                                                  const bf16_t* __restrict__ xtb,
                                                  bf16_t* __restrict__ ctx) {
  __shared__ __align__(16) bf16_t ldsP[16 * 32];
  const int lane = threadIdx.x;
  const int mb = blockIdx.x, h = blockIdx.y, b = blockIdx.z;
  const bf16_t* xrow  = xb  + (size_t)b * XB_ROWS * Dd;   // keys (padded rows)
  const bf16_t* xtrow = xtb + (size_t)b * Dd * XT_LD;     // keys^T (padded cols)
  const v16bf qa0 = load_tile_a(qb, Dd, mb * 16, h * HDd,      lane);
  const v16bf qa1 = load_tile_a(qb, Dd, mb * 16, h * HDd + 32, lane);
  v8f o0 = {}, o1 = {}, o2 = {}, o3 = {};
  float psum[8];
  #pragma unroll
  for (int j = 0; j < 8; ++j) psum[j] = 0.f;
  const int keyc = lane & 15;
  const int hi8 = (lane >> 4) << 3;

  for (int nb = 0; nb < Nn; nb += 32) {
    // S = Q (16x64) . K^T -> two 16x16 f32 tiles
    v8f s0 = {}, s1 = {};
    {
      v16bf kb = load_tile_b(xrow, Dd, nb,      h * HDd,      lane);
      s0 = wmma_bf16(qa0, kb, s0);
      kb = load_tile_b(xrow, Dd, nb,      h * HDd + 32, lane);
      s0 = wmma_bf16(qa1, kb, s0);
      kb = load_tile_b(xrow, Dd, nb + 16, h * HDd,      lane);
      s1 = wmma_bf16(qa0, kb, s1);
      kb = load_tile_b(xrow, Dd, nb + 16, h * HDd + 32, lane);
      s1 = wmma_bf16(qa1, kb, s1);
    }
    // P = exp(S/8) with key-tail masking (branch-free select)
    const bool v0 = (nb + keyc) < Nn;
    const bool v1 = (nb + 16 + keyc) < Nn;
    #pragma unroll
    for (int j = 0; j < 8; ++j) {
      float p0 = v0 ? __expf(s0[j] * 0.125f) : 0.f;
      float p1 = v1 ? __expf(s1[j] * 0.125f) : 0.f;
      psum[j] += p0 + p1;
      ldsP[(j + hi8) * 32 + keyc]      = (bf16_t)p0;   // P row-major 16x32
      ldsP[(j + hi8) * 32 + 16 + keyc] = (bf16_t)p1;
    }
    __syncthreads();
    // O += P (16x32) . K (32x64)
    v16bf pa = load_tile_a(ldsP, 32, 0, 0, lane);
    v16bf kb = load_tile_b(xtrow, XT_LD, h * HDd +  0, nb, lane);
    o0 = wmma_bf16(pa, kb, o0);
    kb = load_tile_b(xtrow, XT_LD, h * HDd + 16, nb, lane);
    o1 = wmma_bf16(pa, kb, o1);
    kb = load_tile_b(xtrow, XT_LD, h * HDd + 32, nb, lane);
    o2 = wmma_bf16(pa, kb, o2);
    kb = load_tile_b(xtrow, XT_LD, h * HDd + 48, nb, lane);
    o3 = wmma_bf16(pa, kb, o3);
    __syncthreads();
  }
  // Row-sum reduce across each 16-lane half (wave32), normalize, store ctx bf16.
  #pragma unroll
  for (int j = 0; j < 8; ++j) {
    float s = psum[j];
    s += __shfl_xor(s, 1, 32);
    s += __shfl_xor(s, 2, 32);
    s += __shfl_xor(s, 4, 32);
    s += __shfl_xor(s, 8, 32);
    psum[j] = 1.0f / s;
  }
  const size_t rbase = (size_t)b * Mm + mb * 16;
  #pragma unroll
  for (int j = 0; j < 8; ++j) {
    size_t row = rbase + j + hi8;
    bf16_t* dst = ctx + row * Dd + h * HDd + keyc;
    dst[0]  = (bf16_t)(o0[j] * psum[j]);
    dst[16] = (bf16_t)(o1[j] * psum[j]);
    dst[32] = (bf16_t)(o2[j] * psum[j]);
    dst[48] = (bf16_t)(o3[j] * psum[j]);
  }
}

// -------------------------------------------------------------------------
// bf16 WMMA GEMM: C = A (MxK) * W^T-stored (padded NxK) + bias, fused epilogues.
// MODE 0: bias -> f32 out; MODE 1: bias+GELU -> bf16 out;
// MODE 2: bias+residual -> f32 out.
// Block = 128 threads (4 waves); block tile = 128x64; each wave owns 32x64.
// The shared 64x32 B^T panel is staged into LDS once per k-step with async
// global->LDS loads (ASYNCcnt), then all 4 waves read B tiles from LDS.
// M must be a multiple of 128; B rows (N) must be padded to block coverage.
// -------------------------------------------------------------------------
template <int MODE>
__global__ __launch_bounds__(128) void gemm_kernel(const bf16_t* __restrict__ A, int lda,
                                                   const bf16_t* __restrict__ Bt, int ldb,
                                                   int Nc, int Kk,
                                                   const float* __restrict__ bias,
                                                   const float* __restrict__ resid, int ldr,
                                                   float* __restrict__ outF,
                                                   bf16_t* __restrict__ outB, int ldo) {
  __shared__ __align__(16) bf16_t ldsB[64 * 32];   // B^T panel: 64 n-rows x 32 k
  const int tid  = threadIdx.x;
  const int lane = tid & 31;
  const int wave = tid >> 5;
  const int row0 = blockIdx.y * 128 + wave * 32;
  const int col0 = blockIdx.x * 64;
  // staging map: 64 rows x 64B; each thread copies 2 x 16B of one row
  const int srow = tid >> 1;
  const int soff = (tid & 1) << 3;                 // 0 or 8 bf16
  const bf16_t* sg = Bt + (size_t)(col0 + srow) * ldb + soff;
  bf16_t* sl = ldsB + srow * 32 + soff;

  v8f acc0 = {}, acc1 = {}, acc2 = {}, acc3 = {};
  v8f acc4 = {}, acc5 = {}, acc6 = {}, acc7 = {};
  for (int k = 0; k < Kk; k += 32) {
    async_copy16(sg + k,      sl);
    async_copy16(sg + k + 16, sl + 16);
    async_wait_all();
    __syncthreads();
    v16bf a0 = load_tile_a(A, lda, row0,      k, lane);
    v16bf a1 = load_tile_a(A, lda, row0 + 16, k, lane);
    v16bf b = load_tile_b(ldsB, 32,  0, 0, lane);
    acc0 = wmma_bf16(a0, b, acc0);
    acc4 = wmma_bf16(a1, b, acc4);
    b = load_tile_b(ldsB, 32, 16, 0, lane);
    acc1 = wmma_bf16(a0, b, acc1);
    acc5 = wmma_bf16(a1, b, acc5);
    b = load_tile_b(ldsB, 32, 32, 0, lane);
    acc2 = wmma_bf16(a0, b, acc2);
    acc6 = wmma_bf16(a1, b, acc6);
    b = load_tile_b(ldsB, 32, 48, 0, lane);
    acc3 = wmma_bf16(a0, b, acc3);
    acc7 = wmma_bf16(a1, b, acc7);
    __syncthreads();
  }
  const int hi8 = (lane >> 4) << 3;
  const int cl = lane & 15;
  #pragma unroll
  for (int t = 0; t < 8; ++t) {
    v8f acc;
    switch (t) {
      case 0: acc = acc0; break; case 1: acc = acc1; break;
      case 2: acc = acc2; break; case 3: acc = acc3; break;
      case 4: acc = acc4; break; case 5: acc = acc5; break;
      case 6: acc = acc6; break; default: acc = acc7; break;
    }
    int c = col0 + (t & 3) * 16 + cl;
    if (c >= Nc) continue;
    int rbase = row0 + ((t >> 2) ? 16 : 0) + hi8;
    float bv = bias[c];
    #pragma unroll
    for (int j = 0; j < 8; ++j) {
      int r = rbase + j;
      float v = acc[j] + bv;
      if (MODE == 1) v = 0.5f * v * (1.0f + erff(v * 0.70710678118654752f));
      if (MODE == 2) v += resid[(size_t)r * ldr + c];
      if (MODE == 1) outB[(size_t)r * ldo + c] = (bf16_t)v;
      else           outF[(size_t)r * ldo + c] = v;
    }
  }
}

// -------------------------------------------------------------------------
// LayerNorm over D=1024 (f32 in -> bf16 out)
// -------------------------------------------------------------------------
__global__ __launch_bounds__(256) void ln_kernel(const float* __restrict__ in,
                                                 const float* __restrict__ sc,
                                                 const float* __restrict__ sh,
                                                 float eps, bf16_t* __restrict__ out) {
  __shared__ float red[256];
  const int row = blockIdx.x, tid = threadIdx.x;
  const float* xr = in + (size_t)row * Dd;
  float v0 = xr[tid], v1 = xr[tid + 256], v2 = xr[tid + 512], v3 = xr[tid + 768];
  red[tid] = v0 + v1 + v2 + v3;
  __syncthreads();
  for (int off = 128; off > 0; off >>= 1) {
    if (tid < off) red[tid] += red[tid + off];
    __syncthreads();
  }
  float mean = red[0] * (1.0f / Dd);
  __syncthreads();
  float d0 = v0 - mean, d1 = v1 - mean, d2 = v2 - mean, d3 = v3 - mean;
  red[tid] = d0 * d0 + d1 * d1 + d2 * d2 + d3 * d3;
  __syncthreads();
  for (int off = 128; off > 0; off >>= 1) {
    if (tid < off) red[tid] += red[tid + off];
    __syncthreads();
  }
  float rstd = rsqrtf(red[0] * (1.0f / Dd) + eps);
  bf16_t* orow = out + (size_t)row * Dd;
  orow[tid]       = (bf16_t)(d0 * rstd * sc[tid]       + sh[tid]);
  orow[tid + 256] = (bf16_t)(d1 * rstd * sc[tid + 256] + sh[tid + 256]);
  orow[tid + 512] = (bf16_t)(d2 * rstd * sc[tid + 512] + sh[tid + 512]);
  orow[tid + 768] = (bf16_t)(d3 * rstd * sc[tid + 768] + sh[tid + 768]);
}

// -------------------------------------------------------------------------
extern "C" void kernel_launch(void* const* d_in, const int* in_sizes, int n_in,
                              void* d_out, int out_size, void* d_ws, size_t ws_size,
                              hipStream_t stream) {
  const float* x      = (const float*)d_in[0];
  const float* pos    = (const float*)d_in[1];
  const int*   idx    = (const int*)d_in[2];
  const float* q      = (const float*)d_in[3];
  const float* proj_w = (const float*)d_in[4];
  const float* proj_b = (const float*)d_in[5];
  const float* ln1s   = (const float*)d_in[6];
  const float* ln1b   = (const float*)d_in[7];
  const float* fc1w   = (const float*)d_in[8];
  const float* fc1b   = (const float*)d_in[9];
  const float* fc2w   = (const float*)d_in[10];
  const float* fc2b   = (const float*)d_in[11];
  const float* ln2s   = (const float*)d_in[12];
  const float* ln2b   = (const float*)d_in[13];
  const float* headw  = (const float*)d_in[14];
  const float* headb  = (const float*)d_in[15];
  float* out = (float*)d_out;
  int* outi = (int*)d_out;
  (void)in_sizes; (void)n_in; (void)out_size; (void)ws_size;

  char* ws = (char*)d_ws;
  size_t off = 0;
  auto take = [&](size_t bytes) -> char* {
    char* p = ws + off;
    off = (off + bytes + 255) & ~(size_t)255;
    return p;
  };
  bf16_t* qbf    = (bf16_t*)take((size_t)Mm * Dd * 2);
  bf16_t* xbf    = (bf16_t*)take((size_t)Bsz * XB_ROWS * Dd * 2);
  bf16_t* xtbf   = (bf16_t*)take((size_t)Bsz * Dd * XT_LD * 2);
  bf16_t* projT  = (bf16_t*)take((size_t)Dd * Dd * 2);
  bf16_t* fc1T   = (bf16_t*)take((size_t)HID * Dd * 2);
  bf16_t* fc2T   = (bf16_t*)take((size_t)Dd * HID * 2);
  bf16_t* headT  = (bf16_t*)take((size_t)HEAD_NPAD * Dd * 2);
  bf16_t* ctx    = (bf16_t*)take((size_t)ROWS * Dd * 2);
  float*  xaggr  = (float*)take((size_t)ROWS * Dd * 4);
  bf16_t* ln1o   = (bf16_t*)take((size_t)ROWS * Dd * 2);
  bf16_t* h1     = (bf16_t*)take((size_t)ROWS * HID * 2);
  float*  xaggr2 = (float*)take((size_t)ROWS * Dd * 4);
  bf16_t* ln2o   = (bf16_t*)take((size_t)ROWS * Dd * 2);
  float*  qsum   = (float*)take((size_t)Dd * 4);
  float*  sc     = (float*)take((size_t)Bsz * Nn * 4);
  int*    order  = (int*)take((size_t)Bsz * Nn * 4);

  // ---- prep: bf16 conversions & transposes (all padded regions zero-filled) ----
  { size_t tot = (size_t)XB_ROWS * Dd;
    dim3 g((unsigned)((tot + 255) / 256), 1, Bsz);
    cvt_pad_kernel<<<g, 256, 0, stream>>>(x, xbf, Nn, XB_ROWS, Dd); }
  { size_t n = (size_t)Mm * Dd;
    cvt_kernel<<<(unsigned)((n + 255) / 256), 256, 0, stream>>>(q, qbf, n); }
  { dim3 g(XT_LD / 16, Dd / 16, Bsz), bl(16, 16);
    tr_cvt_kernel<<<g, bl, 0, stream>>>(x, xtbf, Nn, Dd, XT_LD, XT_LD, Dd,
                                        (size_t)Nn * Dd, (size_t)Dd * XT_LD); }
  { dim3 g(Dd / 16, Dd / 16, 1), bl(16, 16);
    tr_cvt_kernel<<<g, bl, 0, stream>>>(proj_w, projT, Dd, Dd, Dd, Dd, Dd, 0, 0); }
  { dim3 g(Dd / 16, HID / 16, 1), bl(16, 16);
    tr_cvt_kernel<<<g, bl, 0, stream>>>(fc1w, fc1T, Dd, HID, Dd, Dd, HID, 0, 0); }
  { dim3 g(HID / 16, Dd / 16, 1), bl(16, 16);
    tr_cvt_kernel<<<g, bl, 0, stream>>>(fc2w, fc2T, HID, Dd, HID, HID, Dd, 0, 0); }
  { dim3 g(Dd / 16, HEAD_NPAD / 16, 1), bl(16, 16);
    tr_cvt_kernel<<<g, bl, 0, stream>>>(headw, headT, Dd, Cc, Dd, Dd, HEAD_NPAD, 0, 0); }

  // ---- scores / sort / gathers (independent of GEMM chain) ----
  qsum_kernel<<<Dd / 256, 256, 0, stream>>>(q, qsum);
  { dim3 g(Nn, Bsz); scores_kernel<<<g, 256, 0, stream>>>(x, qsum, sc); }
  sort_kernel<<<Bsz, 1024, 0, stream>>>(sc, order);
  gather_x_kernel<<<Bsz * Nn, 256, 0, stream>>>(x, order, out);
  gather_pos_kernel<<<Bsz * (Nn - 1), 256, 0, stream>>>(pos, order, out);
  gather_idx_kernel<<<(Bsz * Nn + 255) / 256, 256, 0, stream>>>(idx, order, outi);

  // ---- attention ----
  { dim3 g(Mm / 16, Hh, Bsz);
    attn_kernel<<<g, 32, 0, stream>>>(qbf, xbf, xtbf, ctx); }

  // ---- projection + MLP + head ----
  { dim3 g(Dd / 64, ROWS / 128);
    gemm_kernel<0><<<g, 128, 0, stream>>>(ctx, Dd, projT, Dd, Dd, Dd,
                                          proj_b, nullptr, 0, xaggr, nullptr, Dd); }
  ln_kernel<<<ROWS, 256, 0, stream>>>(xaggr, ln1s, ln1b, 1e-6f, ln1o);
  { dim3 g(HID / 64, ROWS / 128);
    gemm_kernel<1><<<g, 128, 0, stream>>>(ln1o, Dd, fc1T, Dd, HID, Dd,
                                          fc1b, nullptr, 0, nullptr, h1, HID); }
  { dim3 g(Dd / 64, ROWS / 128);
    gemm_kernel<2><<<g, 128, 0, stream>>>(h1, HID, fc2T, HID, Dd, HID,
                                          fc2b, xaggr, Dd, xaggr2, nullptr, Dd); }
  ln_kernel<<<ROWS, 256, 0, stream>>>(xaggr2, ln2s, ln2b, 1e-5f, ln2o);
  { dim3 g(HEAD_NPAD / 64, ROWS / 128);
    gemm_kernel<0><<<g, 128, 0, stream>>>(ln2o, Dd, headT, Dd, Cc, Dd,
                                          headb, nullptr, 0, out + O_PRED, nullptr, Cc); }
}
